// EncoderBlock_80504866996513
// MI455X (gfx1250) — compile-verified, matrix operand order checked by reading, not run on hardware
//
#include <hip/hip_runtime.h>
#include <hip/hip_bf16.h>

// ---------------------------------------------------------------------------
// Encoder block: LN -> QKV -> softmax(QK^T)*(1/sqrt(D)) -> PV -> MLP(swish)
// B=8, S=2048, D=1024, H=4096.
// GEMMs: v_wmma_f32_16x16x32_f16, A tile via global_load_async_to_lds_b128
// (ASYNCcnt), B tile via Tensor Data Mover tensor_load_to_lds (TENSORcnt),
// explicit double buffering.
// ---------------------------------------------------------------------------

typedef _Float16 half_t;
typedef __attribute__((ext_vector_type(16))) _Float16 v16h;
typedef __attribute__((ext_vector_type(8)))  _Float16 v8h;
typedef __attribute__((ext_vector_type(8)))  float    v8f;
typedef __attribute__((ext_vector_type(4)))  unsigned int v4u;
typedef __attribute__((ext_vector_type(8)))  unsigned int v8u;

static constexpr int  Bc = 8;
static constexpr int  Sc = 2048;
static constexpr int  Dc = 1024;
static constexpr int  Hc = 4096;
static constexpr int  Mc = Bc * Sc;          // 16384 rows
static constexpr int  BM = 128, BN = 64, BK = 32;
static constexpr int  LDS_STRIDE = 40;       // 32 halves + 8 pad = 80B rows (16B aligned)
static constexpr int  A_TILE_H = BM * LDS_STRIDE;  // halves per A buffer
static constexpr int  B_TILE_H = BN * LDS_STRIDE;  // halves per B buffer

// ---------------------------------------------------------------------------
// Wave / block reductions (wave32)
// ---------------------------------------------------------------------------
__device__ __forceinline__ float wave_sum(float v) {
#pragma unroll
  for (int o = 16; o > 0; o >>= 1) v += __shfl_xor(v, o, 32);
  return v;
}
__device__ __forceinline__ float wave_max(float v) {
#pragma unroll
  for (int o = 16; o > 0; o >>= 1) v = fmaxf(v, __shfl_xor(v, o, 32));
  return v;
}
__device__ __forceinline__ float block_sum(float v, float* sh) {
  v = wave_sum(v);
  const int w = threadIdx.x >> 5, l = threadIdx.x & 31;
  if (l == 0) sh[w] = v;
  __syncthreads();
  float t = 0.f;
#pragma unroll
  for (int i = 0; i < 8; ++i) t += sh[i];
  __syncthreads();
  return t;
}
__device__ __forceinline__ float block_max(float v, float* sh) {
  v = wave_max(v);
  const int w = threadIdx.x >> 5, l = threadIdx.x & 31;
  if (l == 0) sh[w] = v;
  __syncthreads();
  float t = -3.4e38f;
#pragma unroll
  for (int i = 0; i < 8; ++i) t = fmaxf(t, sh[i]);
  __syncthreads();
  return t;
}

// ---------------------------------------------------------------------------
// CDNA5 async data movement helpers
// ---------------------------------------------------------------------------
// 16B global -> LDS async copy (tracked by ASYNCcnt)
__device__ __forceinline__ void async_copy16(unsigned lds_off,
                                             unsigned long long gaddr) {
  asm volatile("global_load_async_to_lds_b128 %0, %1, off"
               :: "v"(lds_off), "v"(gaddr)
               : "memory");
}
__device__ __forceinline__ void wait_async(int pending) {
  if (pending)
    asm volatile("s_wait_asynccnt 0x2" ::: "memory");
  else
    asm volatile("s_wait_asynccnt 0x0" ::: "memory");
}
__device__ __forceinline__ void wait_tensor(int pending) {
  if (pending)
    __builtin_amdgcn_s_wait_tensorcnt(1);
  else
    __builtin_amdgcn_s_wait_tensorcnt(0);
}

// Tensor Data Mover: DMA one BN x BK f16 tile (row stride = ld elements) from
// global memory into LDS at lds_byte_off, writing rows with LDS_STRIDE layout
// (64B of data + 16B pad -> pad_interval=3 (16 DWORDs), pad_amount=3 (4 DWORDs)).
// 2-D tensor: only D# groups 0 and 1. Tracked by TENSORcnt.
__device__ __forceinline__ void tdm_load_tile_f16(unsigned lds_byte_off,
                                                  unsigned long long gaddr,
                                                  unsigned long long ld) {
  v4u g0;
  g0[0] = 1u;                                        // count=1, user descriptor
  g0[1] = lds_byte_off;                              // lds_addr
  g0[2] = (unsigned)gaddr;                           // global_addr[31:0]
  g0[3] = (unsigned)(gaddr >> 32) | (2u << 30);      // global_addr[56:32] | type=2

  v8u g1;
  const unsigned td0 = (unsigned)ld;                 // tensor_dim0 (row length)
  g1[0] = (1u << 16)                                 // data_size = 2 bytes
        | (1u << 20)                                 // pad_enable
        | (3u << 22)                                 // pad_interval: 16 DWORDs (64B)
        | (3u << 25);                                // pad_amount: 4 DWORDs (16B)
  g1[1] = (td0 & 0xFFFFu) << 16;                     // tensor_dim0[15:0]
  g1[2] = (td0 >> 16) | ((unsigned)BN << 16);        // tensor_dim0[31:16] | tensor_dim1[15:0]
  g1[3] = ((unsigned)BK << 16);                      // tensor_dim1[31:16]=0 | tile_dim0=BK
  g1[4] = (unsigned)BN;                              // tile_dim1=BN, tile_dim2=0
  g1[5] = (unsigned)(ld & 0xFFFFFFFFull);            // tensor_dim0_stride[31:0]
  g1[6] = (unsigned)((ld >> 32) & 0xFFFFull);        // tensor_dim0_stride[47:32]
  g1[7] = 0u;                                        // tensor_dim1_stride (unused, 2D)

  asm volatile("tensor_load_to_lds %0, %1"
               :: "s"(g0), "s"(g1)
               : "memory");
}

// ---------------------------------------------------------------------------
// fp32 -> f16 transpose:  out[c*R + r] = in[r*C + c]
// ---------------------------------------------------------------------------
__global__ __launch_bounds__(256) void convT_kernel(const float* __restrict__ in,
                                                    int R, int C,
                                                    half_t* __restrict__ out) {
  long idx = (long)blockIdx.x * 256 + threadIdx.x;
  if (idx >= (long)R * C) return;
  long r = idx / C, c = idx % C;
  out[c * (long)R + r] = (half_t)in[idx];
}

// ---------------------------------------------------------------------------
// LayerNorm: one block per row of D=1024; f32 in, f16 out
// ---------------------------------------------------------------------------
__global__ __launch_bounds__(256) void ln_kernel(const float* __restrict__ x,
                                                 const float* __restrict__ gamma,
                                                 const float* __restrict__ beta,
                                                 half_t* __restrict__ xn) {
  __shared__ float sred[8];
  const long row = blockIdx.x;
  const float* xr = x + row * (long)Dc;
  float v[4];
  float s = 0.f;
#pragma unroll
  for (int i = 0; i < 4; ++i) {
    v[i] = xr[i * 256 + threadIdx.x];
    s += v[i];
  }
  const float mu = block_sum(s, sred) * (1.0f / Dc);
  float vs = 0.f;
#pragma unroll
  for (int i = 0; i < 4; ++i) {
    float d = v[i] - mu;
    vs += d * d;
  }
  const float var  = block_sum(vs, sred) * (1.0f / Dc);
  const float rstd = rsqrtf(var + 1e-5f);
#pragma unroll
  for (int i = 0; i < 4; ++i) {
    int j = i * 256 + threadIdx.x;
    xn[row * (long)Dc + j] = (half_t)((v[i] - mu) * rstd * gamma[j] + beta[j]);
  }
}

// ---------------------------------------------------------------------------
// Row softmax over S=2048 f16 scores; fold in 1/sqrt(D)=1/32 (applied AFTER
// softmax per the reference); f16 probs out.
// ---------------------------------------------------------------------------
__global__ __launch_bounds__(256) void softmax_kernel(const half_t* __restrict__ scores,
                                                      half_t* __restrict__ probs) {
  __shared__ float sred[8];
  const long row = blockIdx.x;
  const half_t* sr = scores + row * (long)Sc;
  float v[8];
  float m = -3.4e38f;
#pragma unroll
  for (int i = 0; i < 8; ++i) {
    v[i] = (float)sr[i * 256 + threadIdx.x];
    m = fmaxf(m, v[i]);
  }
  m = block_max(m, sred);
  float s = 0.f;
#pragma unroll
  for (int i = 0; i < 8; ++i) {
    v[i] = __expf(v[i] - m);
    s += v[i];
  }
  s = block_sum(s, sred);
  const float inv = (1.0f / 32.0f) / s;   // 1/sqrt(1024) == 1/32
#pragma unroll
  for (int i = 0; i < 8; ++i)
    probs[row * (long)Sc + i * 256 + threadIdx.x] = (half_t)(v[i] * inv);
}

// ---------------------------------------------------------------------------
// A-fragment (16x32 f16) per-lane load from LDS (also used for B of NT GEMM):
// lane l: row M=l%16, K = kb..kb+7 and kb+16..kb+23 with kb = (l/16)*8
// ---------------------------------------------------------------------------
__device__ __forceinline__ v16h load_frag(const half_t* p, int kb) {
  union { v16h v; v8h h[2]; } u;
  u.h[0] = *(const v8h*)(p + kb);
  u.h[1] = *(const v8h*)(p + kb + 16);
  return u.v;
}

// ---------------------------------------------------------------------------
// NT GEMM: C[M,N] = A[M,K] * Bt[N,K]^T, f16 in / f32 accum, 128x64 block tile,
// 8 waves of 32x32 each (2x2 v_wmma_f32_16x16x32_f16), BK=32, double-buffered:
// A tile via async-to-LDS (ASYNCcnt), B tile via TDM (TENSORcnt).
// EPI: 0 = store f16 C[m,n]
//      1 = store f16 transposed C -> out[n*ldc + m]   (produces V^T)
//      2 = f16( swish(c + bias[n]) )
//      3 = f32( c + bias[n] + resid[m,n] )            (final output)
// ---------------------------------------------------------------------------
template <int EPI>
__global__ __launch_bounds__(256) void gemm_nt(
    const half_t* __restrict__ A,  long lda, long sAz,
    const half_t* __restrict__ Bt, long ldb, long sBz,
    void* __restrict__ Cout,       long ldc, long sCz,
    int K,
    const float* __restrict__ bias,
    const float* __restrict__ resid, long ldres) {
  __shared__ half_t shA[2 * A_TILE_H];
  __shared__ half_t shB[2 * B_TILE_H];

  const int tid  = threadIdx.x;
  const int lane = tid & 31;
  const int wave = tid >> 5;
  const int wm   = wave & 3;        // 4 waves along M (4*32 = 128)
  const int wn   = wave >> 2;       // 2 waves along N (2*32 = 64)
  const int z    = blockIdx.z;
  const long tileM = (long)blockIdx.y * BM;
  const long tileN = (long)blockIdx.x * BN;

  A  += (long)z * sAz;
  Bt += (long)z * sBz;

  const unsigned shA_base = (unsigned)(unsigned long long)(uintptr_t)&shA[0];
  const unsigned shB_base = (unsigned)(unsigned long long)(uintptr_t)&shB[0];

  const int lm = lane & 15;
  const int kb = (lane >> 4) << 3;
  const int hi = lane >> 4;

  // my two A-tile chunks (row, 16B column chunk)
  const int ar0 = tid >> 2,           ac0 = (tid & 3) << 3;
  const int ar1 = (tid + 256) >> 2,   ac1 = ac0;

  const int ktiles = K >> 5;

  auto issueA = [&](int buf, int kt) {
    const long k0 = (long)kt << 5;
    const unsigned base = shA_base + (unsigned)(buf * A_TILE_H) * 2u;
    async_copy16(base + (unsigned)(ar0 * LDS_STRIDE + ac0) * 2u,
                 (unsigned long long)(uintptr_t)&A[(tileM + ar0) * lda + k0 + ac0]);
    async_copy16(base + (unsigned)(ar1 * LDS_STRIDE + ac1) * 2u,
                 (unsigned long long)(uintptr_t)&A[(tileM + ar1) * lda + k0 + ac1]);
  };
  auto issueB = [&](int buf, int kt) {   // one TDM op covers the whole B tile
    const long k0 = (long)kt << 5;
    tdm_load_tile_f16(shB_base + (unsigned)(buf * B_TILE_H) * 2u,
                      (unsigned long long)(uintptr_t)&Bt[tileN * ldb + k0],
                      (unsigned long long)ldb);
  };

  // prologue: fill both pipeline stages
  issueA(0, 0);
  if (wave == 0) issueB(0, 0);
  if (ktiles > 1) {
    issueA(1, 1);
    if (wave == 0) issueB(1, 1);
  }

  v8f acc[2][2] = {};

  for (int kt = 0; kt < ktiles; ++kt) {
    const int buf = kt & 1;
    const int more = (kt + 1 < ktiles);
    wait_async(more);                                     // oldest A batch done
    if (wave == 0) wait_tensor(more);                     // oldest B TDM done
    __syncthreads();                                      // tile published to all waves

    const half_t* tA = &shA[buf * A_TILE_H];
    const half_t* tB = &shB[buf * B_TILE_H];
    v16h a0 = load_frag(&tA[(wm * 32 +      lm) * LDS_STRIDE], kb);
    v16h a1 = load_frag(&tA[(wm * 32 + 16 + lm) * LDS_STRIDE], kb);
    v16h b0 = load_frag(&tB[(wn * 32 +      lm) * LDS_STRIDE], kb);
    v16h b1 = load_frag(&tB[(wn * 32 + 16 + lm) * LDS_STRIDE], kb);

    acc[0][0] = __builtin_amdgcn_wmma_f32_16x16x32_f16(false, a0, false, b0,
                                                       (short)0, acc[0][0], false, false);
    acc[0][1] = __builtin_amdgcn_wmma_f32_16x16x32_f16(false, a0, false, b1,
                                                       (short)0, acc[0][1], false, false);
    acc[1][0] = __builtin_amdgcn_wmma_f32_16x16x32_f16(false, a1, false, b0,
                                                       (short)0, acc[1][0], false, false);
    acc[1][1] = __builtin_amdgcn_wmma_f32_16x16x32_f16(false, a1, false, b1,
                                                       (short)0, acc[1][1], false, false);

    __syncthreads();                                      // everyone done reading buf
    if (kt + 2 < ktiles) {
      issueA(buf, kt + 2);                                // refill freed buffer
      if (wave == 0) issueB(buf, kt + 2);
      // warm L2 a bit further ahead on the streaming (A) operand
      __builtin_prefetch(&A[(tileM + ar0) * lda + ((long)(kt + 4) << 5)], 0, 3);
    }
  }

  // Epilogue. C layout: VGPR r, lane l -> M = r + (l/16)*8, N = l%16
#pragma unroll
  for (int fm = 0; fm < 2; ++fm) {
#pragma unroll
    for (int fn = 0; fn < 2; ++fn) {
#pragma unroll
      for (int r = 0; r < 8; ++r) {
        const long m = tileM + wm * 32 + fm * 16 + r + hi * 8;
        const long n = tileN + wn * 32 + fn * 16 + lm;
        const float cv = acc[fm][fn][r];
        if (EPI == 0) {
          ((half_t*)Cout)[(long)z * sCz + m * ldc + n] = (half_t)cv;
        } else if (EPI == 1) {
          ((half_t*)Cout)[(long)z * sCz + n * ldc + m] = (half_t)cv;
        } else if (EPI == 2) {
          const float t = cv + bias[n];
          ((half_t*)Cout)[(long)z * sCz + m * ldc + n] =
              (half_t)(t / (1.0f + __expf(-t)));   // swish
        } else {
          const float t = cv + bias[n] + resid[m * ldres + n];
          ((float*)Cout)[m * ldc + n] = t;
        }
      }
    }
  }
}

// ---------------------------------------------------------------------------
// Host-side orchestration
// ---------------------------------------------------------------------------
extern "C" void kernel_launch(void* const* d_in, const int* in_sizes, int n_in,
                              void* d_out, int out_size, void* d_ws, size_t ws_size,
                              hipStream_t stream) {
  (void)in_sizes; (void)n_in; (void)out_size; (void)ws_size;

  const float* x       = (const float*)d_in[0];
  const float* ln_g    = (const float*)d_in[1];
  const float* ln_b    = (const float*)d_in[2];
  const float* Wq      = (const float*)d_in[3];
  const float* Wk      = (const float*)d_in[4];
  const float* Wv      = (const float*)d_in[5];
  const float* W1      = (const float*)d_in[6];
  const float* b1      = (const float*)d_in[7];
  const float* W2      = (const float*)d_in[8];
  const float* b2      = (const float*)d_in[9];
  float*       out     = (float*)d_out;

  // workspace carve-up (f16 buffers), 256B aligned
  char* ws = (char*)d_ws;
  size_t off = 0;
  auto carve = [&](size_t bytes) {
    size_t o = off;
    off = (off + bytes + 255) & ~(size_t)255;
    return o;
  };
  half_t* xn   = (half_t*)(ws + carve((size_t)Mc * Dc * 2));
  half_t* WqT  = (half_t*)(ws + carve((size_t)Dc * Dc * 2));
  half_t* WkT  = (half_t*)(ws + carve((size_t)Dc * Dc * 2));
  half_t* WvT  = (half_t*)(ws + carve((size_t)Dc * Dc * 2));
  half_t* W1T  = (half_t*)(ws + carve((size_t)Dc * Hc * 2));
  half_t* W2T  = (half_t*)(ws + carve((size_t)Hc * Dc * 2));
  half_t* q    = (half_t*)(ws + carve((size_t)Mc * Dc * 2));
  half_t* k    = (half_t*)(ws + carve((size_t)Mc * Dc * 2));
  half_t* vT   = (half_t*)(ws + carve((size_t)Mc * Dc * 2));   // [B][D][S]
  half_t* sc   = (half_t*)(ws + carve((size_t)Bc * Sc * Sc * 2));
  half_t* pr   = (half_t*)(ws + carve((size_t)Bc * Sc * Sc * 2));
  half_t* ao   = (half_t*)(ws + carve((size_t)Mc * Dc * 2));
  half_t* h    = (half_t*)(ws + carve((size_t)Mc * Hc * 2));

  // 1) weight convert + transpose (fp32 -> f16, [K,N] -> [N,K])
  {
    int n1 = Dc * Dc, n2 = Dc * Hc;
    convT_kernel<<<dim3((n1 + 255) / 256), 256, 0, stream>>>(Wq, Dc, Dc, WqT);
    convT_kernel<<<dim3((n1 + 255) / 256), 256, 0, stream>>>(Wk, Dc, Dc, WkT);
    convT_kernel<<<dim3((n1 + 255) / 256), 256, 0, stream>>>(Wv, Dc, Dc, WvT);
    convT_kernel<<<dim3((n2 + 255) / 256), 256, 0, stream>>>(W1, Dc, Hc, W1T);
    convT_kernel<<<dim3((n2 + 255) / 256), 256, 0, stream>>>(W2, Hc, Dc, W2T);
  }

  // 2) LayerNorm
  ln_kernel<<<dim3(Mc), 256, 0, stream>>>(x, ln_g, ln_b, xn);

  // 3) Q = xn*Wq, K = xn*Wk  (f16 out)
  gemm_nt<0><<<dim3(Dc / 64, Mc / 128, 1), 256, 0, stream>>>(
      xn, Dc, 0, WqT, Dc, 0, q, Dc, 0, Dc, nullptr, nullptr, 0);
  gemm_nt<0><<<dim3(Dc / 64, Mc / 128, 1), 256, 0, stream>>>(
      xn, Dc, 0, WkT, Dc, 0, k, Dc, 0, Dc, nullptr, nullptr, 0);
  // V stored transposed per batch: vT[b][n][s]
  gemm_nt<1><<<dim3(Dc / 64, Sc / 128, Bc), 256, 0, stream>>>(
      xn, Dc, (long)Sc * Dc, WvT, Dc, 0, vT, Sc, (long)Dc * Sc, Dc,
      nullptr, nullptr, 0);

  // 4) scores = q * k^T per batch (f16)
  gemm_nt<0><<<dim3(Sc / 64, Sc / 128, Bc), 256, 0, stream>>>(
      q, Dc, (long)Sc * Dc, k, Dc, (long)Sc * Dc, sc, Sc, (long)Sc * Sc, Dc,
      nullptr, nullptr, 0);

  // 5) probs = softmax(scores) * (1/sqrt(D))
  softmax_kernel<<<dim3(Bc * Sc), 256, 0, stream>>>(sc, pr);

  // 6) attnout = probs * V  (Bt = vT is already [N=D, K=S])
  gemm_nt<0><<<dim3(Dc / 64, Sc / 128, Bc), 256, 0, stream>>>(
      pr, Sc, (long)Sc * Sc, vT, Sc, (long)Dc * Sc, ao, Dc, (long)Sc * Dc, Sc,
      nullptr, nullptr, 0);

  // 7) h = swish(attnout * W1 + b1)
  gemm_nt<2><<<dim3(Hc / 64, Mc / 128, 1), 256, 0, stream>>>(
      ao, Dc, 0, W1T, Dc, 0, h, Hc, 0, Dc, b1, nullptr, 0);

  // 8) y = h * W2 + b2 + x  (fp32 out)
  gemm_nt<3><<<dim3(Dc / 64, Mc / 128, 1), 256, 0, stream>>>(
      h, Hc, 0, W2T, Hc, 0, out, Dc, 0, Hc, b2, x, Dc);
}
